// InvariantPointAttention_57896159150737
// MI455X (gfx1250) — matrix operand order repaired and checked by reading, not compiled, not run on hardware
//
#include <hip/hip_runtime.h>

typedef __attribute__((ext_vector_type(16))) __bf16 v16bf;
typedef __attribute__((ext_vector_type(2)))  __bf16 v2bf;
typedef __attribute__((ext_vector_type(8)))  float  v8f;

#define NRES 512
#define SDIM 384
#define NH   12
#define EDIM 16
#define NPQ  4
#define NPV  8
#define CDIM 128
#define FDIM 2112            // NH*(CDIM+EDIM+NPV*4)
#define NN   (NRES*NRES)
#define KT_F (FDIM/32)       // 66 k-tiles in final GEMM

// ---------- helpers ----------

__device__ inline __bf16 bf(float f) { return (__bf16)f; }   // native cvt

__device__ inline unsigned pk2(float a, float b) {           // one v_cvt_pk_bf16_f32
  v2bf v = { (__bf16)a, (__bf16)b };
  union { v2bf v; unsigned u; } p; p.v = v;
  return p.u;
}
// 4 consecutive swizzled elements -> one 8B store
__device__ inline void st4(__bf16* dst, float a, float b, float c, float d) {
  uint2 v; v.x = pk2(a, b); v.y = pk2(c, d);
  *(uint2*)dst = v;
}
// 8 consecutive swizzled elements -> one 16B store
__device__ inline void st8(__bf16* dst, const float* w) {
  uint4 v;
  v.x = pk2(w[0], w[1]); v.y = pk2(w[2], w[3]);
  v.z = pk2(w[4], w[5]); v.w = pk2(w[6], w[7]);
  *(uint4*)dst = v;
}

__device__ inline v8f wmma_bf16(v16bf a, v16bf b, v8f c) {
  return __builtin_amdgcn_wmma_f32_16x16x32_bf16(false, a, false, b, (short)0, c,
                                                 false, false);
}

// Fragment-order addressing within one 512-element 16x32 tile.
// A tile (MxK=16x32): lane = m + 16*k[3]; elem = k[2:0] | (k[4] -> bit3)
__device__ inline int swzA(int m, int k) {
  return (m + ((k & 8) << 1)) * 16 + ((k & 7) | ((k & 16) >> 1));
}
// B tile (KxN=32x16): lane = n + 16*k[3]; elem = k[2:0] | (k[4] -> bit3)
__device__ inline int swzB(int k, int n) {
  return (n + ((k & 8) << 1)) * 16 + ((k & 7) | ((k & 16) >> 1));
}
// att feature buffer: A-operand tiles of the final GEMM: [n-tile][f-tile][512]
__device__ inline size_t attsw(int n, int f) {
  return ((size_t)((n >> 4) * KT_F + (f >> 5))) * 512 + (size_t)swzA(n & 15, f & 31);
}

__device__ inline v16bf frag(const __bf16* p) {       // 32B contiguous load
  return *(const v16bf*)p;
}

// ---------- kernel 1: fused per-head projections ----------
// per head: [512,384] x [384,96] ; cols: q(16) k(16) v(16) qp(12) kp(12) vp(24)
__global__ __launch_bounds__(192)
void k_proj(const float* __restrict__ single,
            const float* __restrict__ Wq, const float* __restrict__ Wk,
            const float* __restrict__ Wv, const float* __restrict__ Wqp,
            const float* __restrict__ Wkp, const float* __restrict__ Wvp,
            float* __restrict__ qb, float* __restrict__ kb, float* __restrict__ vb,
            float* __restrict__ qpb, float* __restrict__ kpb, float* __restrict__ vpb) {
  __shared__ __bf16 As[512];        // one swizzled A tile
  __shared__ __bf16 Bs[6 * 512];    // six swizzled B tiles
  const int tid = threadIdx.x, lane = tid & 31, wv = tid >> 5;
  const int n0 = blockIdx.x * 16, h = blockIdx.y;

  // staging column is invariant per thread: resolve source pointer ONCE
  const int c = tid % 96, kqb = tid / 96;     // kqb in {0,1}
  const float* sptr; int sstr;
  if (c < 16)      { sptr = Wq + (size_t)h * SDIM * EDIM + c;        sstr = EDIM; }
  else if (c < 32) { sptr = Wk + (size_t)h * SDIM * EDIM + (c - 16); sstr = EDIM; }
  else if (c < 48) { sptr = Wv + (size_t)h * SDIM * EDIM + (c - 32); sstr = EDIM; }
  else if (c < 60) { int cc = c - 48, p = cc / 3, j = cc % 3;
                     sptr = Wqp + ((size_t)(h * NPQ + p) * SDIM) * 3 + j; sstr = 3; }
  else if (c < 72) { int cc = c - 60, p = cc / 3, j = cc % 3;
                     sptr = Wkp + ((size_t)(h * NPQ + p) * SDIM) * 3 + j; sstr = 3; }
  else             { int cc = c - 72, p = cc / 3, j = cc % 3;
                     sptr = Wvp + ((size_t)(h * NPV + p) * SDIM) * 3 + j; sstr = 3; }
  __bf16* bdst0 = Bs + (c >> 4) * 512;        // column tile base

  v8f acc = {};
  for (int kt = 0; kt < SDIM / 32; ++kt) {
    const int k0 = kt * 32;
    if (tid < 128) {                // 128 float4 = 16x32 A tile
      int r = tid >> 3, c4 = (tid & 7) * 4;
      float4 s = *(const float4*)(single + (size_t)(n0 + r) * SDIM + k0 + c4);
      st4(As + swzA(r, c4), s.x, s.y, s.z, s.w);
    }
#pragma unroll
    for (int it = 0; it < 4; ++it) {          // branch-free B staging
      int kq = kqb + it * 2;                  // 0..7
      int kk = kq * 4;
      const float* sp = sptr + (size_t)(k0 + kk) * sstr;
      st4(bdst0 + swzB(kk, c & 15), sp[0], sp[sstr], sp[2 * sstr], sp[3 * sstr]);
    }
    __syncthreads();
    acc = wmma_bf16(frag(As + lane * 16), frag(Bs + wv * 512 + lane * 16), acc);
    __syncthreads();
  }

  // epilogue destination is invariant per thread: resolve ONCE, store branch-free
  const int half = lane >> 4, cn = lane & 15;
  const int col = wv * 16 + cn;
  float* dptr; int dstr;
  if (col < 16)      { dptr = qb + (size_t)h * NRES * EDIM + col;        dstr = EDIM; }
  else if (col < 32) { dptr = kb + (size_t)h * NRES * EDIM + (col - 16); dstr = EDIM; }
  else if (col < 48) { dptr = vb + (size_t)h * NRES * EDIM + (col - 32); dstr = EDIM; }
  else if (col < 60) { int cc = col - 48, p = cc / 3, j = cc % 3;
                       dptr = qpb + ((size_t)(h * NPQ + p) * NRES) * 3 + j; dstr = 3; }
  else if (col < 72) { int cc = col - 60, p = cc / 3, j = cc % 3;
                       dptr = kpb + ((size_t)(h * NPQ + p) * NRES) * 3 + j; dstr = 3; }
  else               { int cc = col - 72, p = cc / 3, j = cc % 3;
                       dptr = vpb + ((size_t)(h * NPV + p) * NRES) * 3 + j; dstr = 3; }
#pragma unroll
  for (int r = 0; r < 8; ++r)
    dptr[(size_t)(n0 + r + 8 * half) * dstr] = acc[r];
}

// ---------- kernel 2: frames; emit swizzled logit/value operands ----------
__global__ __launch_bounds__(256)
void k_frames(const float* __restrict__ rot, const float* __restrict__ trans,
              const float* __restrict__ scale_head,
              const float* __restrict__ qb, const float* __restrict__ kb,
              const float* __restrict__ vb, const float* __restrict__ qpb,
              const float* __restrict__ kpb, const float* __restrict__ vpb,
              __bf16* __restrict__ qsw, __bf16* __restrict__ ksw,
              float* __restrict__ mterm, __bf16* __restrict__ vsw,
              float* __restrict__ colsum) {
  int gid = blockIdx.x * blockDim.x + threadIdx.x;
  if (gid >= NH * NRES) return;
  colsum[gid] = 0.f;                      // fresh accumulator every launch
  int h = gid / NRES, n = gid % NRES;
  float R[3][3], t[3];
#pragma unroll
  for (int i = 0; i < 3; ++i) {
    t[i] = trans[n * 3 + i];
#pragma unroll
    for (int j = 0; j < 3; ++j) R[i][j] = rot[n * 9 + i * 3 + j];
  }
  float x = scale_head[h];
  float sp = (x > 20.f) ? x : log1pf(__expf(x));
  float ch = sqrtf(1.f / (18.f * NPQ)) * sp;

  float qrow[32], krow[32];
#pragma unroll
  for (int e = 0; e < 16; ++e) {
    qrow[e] = 0.25f * qb[((size_t)h * NRES + n) * EDIM + e];   // 1/sqrt(E)
    krow[e] = kb[((size_t)h * NRES + n) * EDIM + e];
  }
  float sqk = 0.f;
#pragma unroll
  for (int p = 0; p < NPQ; ++p) {
    float pq[3], pk[3];
#pragma unroll
    for (int j = 0; j < 3; ++j) {
      pq[j] = qpb[(((size_t)h * NPQ + p) * NRES + n) * 3 + j];
      pk[j] = kpb[(((size_t)h * NPQ + p) * NRES + n) * 3 + j];
    }
#pragma unroll
    for (int i = 0; i < 3; ++i) {
      float lq = R[i][0] * pq[0] + R[i][1] * pq[1] + R[i][2] * pq[2] + t[i];
      float lk = R[i][0] * pk[0] + R[i][1] * pk[1] + R[i][2] * pk[2] + t[i];
      qrow[16 + p * 3 + i] = 2.f * ch * lq;   // +2c * q.k cross term
      krow[16 + p * 3 + i] = lk;
      sqk += lk * lk;
    }
  }
#pragma unroll
  for (int e = 28; e < 32; ++e) { qrow[e] = 0.f; krow[e] = 0.f; }
  mterm[gid] = -ch * sqk;                 // -c|k_m|^2 ; -c|q_n|^2 cancels

  // pack fragment rows: each (h,n) owns two 32B runs in qsw and ksw
  const int tile = n >> 4, m = n & 15;
  __bf16* qt = qsw + (size_t)(h * 32 + tile) * 512;
  __bf16* kt = ksw + (size_t)(h * 32 + tile) * 512;
  v16bf vq0, vq1, vk0, vk1;
#pragma unroll
  for (int e = 0; e < 8; ++e) {
    vq0[e] = bf(qrow[e]);       vq0[e + 8] = bf(qrow[16 + e]);  // lane m
    vq1[e] = bf(qrow[8 + e]);   vq1[e + 8] = bf(qrow[24 + e]);  // lane m+16
  }
#pragma unroll
  for (int e = 0; e < 8; ++e) {
    vk0[e] = bf(krow[e]);       vk0[e + 8] = bf(krow[16 + e]);
    vk1[e] = bf(krow[8 + e]);   vk1[e + 8] = bf(krow[24 + e]);
  }
  *(v16bf*)(qt + m * 16) = vq0;
  *(v16bf*)(qt + (m + 16) * 16) = vq1;
  *(v16bf*)(kt + m * 16) = vk0;
  *(v16bf*)(kt + (m + 16) * 16) = vk1;

  // value matrix (B operand of aggregation): scattered b16 stores (small)
  const int vkt = n >> 5, vkk = n & 31;
  __bf16* vt = vsw + ((size_t)(h * 16 + vkt) * 3) * 512;
#pragma unroll
  for (int e = 0; e < 16; ++e)
    vt[swzB(vkk, e)] = bf(vb[((size_t)h * NRES + n) * EDIM + e]);
#pragma unroll
  for (int p = 0; p < NPV; ++p) {
    float pv[3];
#pragma unroll
    for (int j = 0; j < 3; ++j) pv[j] = vpb[(((size_t)h * NPV + p) * NRES + n) * 3 + j];
#pragma unroll
    for (int i = 0; i < 3; ++i) {
      float lv = R[i][0] * pv[0] + R[i][1] * pv[1] + R[i][2] * pv[2] + t[i];
      int cc = 16 + p * 3 + i;
      vt[(cc >> 4) * 512 + swzB(vkk, cc & 15)] = bf(lv);
    }
  }
#pragma unroll
  for (int cc = 40; cc < 48; ++cc)
    vt[(cc >> 4) * 512 + swzB(vkk, cc & 15)] = bf(0.f);
}

// ---------- small conversion kernels (operands to fragment order) ----------
__global__ __launch_bounds__(256)
void k_cvt_wb(const float* __restrict__ Wb, __bf16* __restrict__ wbsw) {
  int i = blockIdx.x * 256 + threadIdx.x;     // (kc, hh) over 128x16
  if (i >= 128 * 16) return;
  int kc = i >> 4, hh = i & 15;
  float v = (hh < NH) ? Wb[hh * CDIM + kc] : 0.f;
  wbsw[(kc >> 5) * 512 + swzB(kc & 31, hh)] = bf(v);
}

__global__ __launch_bounds__(256)
void k_cvt_col(const float* __restrict__ colsum, __bf16* __restrict__ colsw) {
  int i = blockIdx.x * 256 + threadIdx.x;     // (hh, m) over 16x512
  if (i >= 16 * 512) return;
  int hh = i >> 9, m = i & 511;
  float v = (hh < NH) ? colsum[hh * NRES + m] : 0.f;
  colsw[(m >> 5) * 512 + swzA(hh, m & 31)] = bf(v);
}

__global__ __launch_bounds__(256)
void k_cvt_wout(const float* __restrict__ Wout, __bf16* __restrict__ woutsw) {
  int idx = blockIdx.x * 256 + threadIdx.x;   // (f-quad, s) over 528x384
  if (idx >= (FDIM / 4) * SDIM) return;
  int s = idx % SDIM, f = (idx / SDIM) * 4;
  float w0 = Wout[(size_t)(f + 0) * SDIM + s];
  float w1 = Wout[(size_t)(f + 1) * SDIM + s];
  float w2 = Wout[(size_t)(f + 2) * SDIM + s];
  float w3 = Wout[(size_t)(f + 3) * SDIM + s];
  st4(woutsw + ((size_t)(f >> 5) * 24 + (s >> 4)) * 512 + swzB(f & 31, s & 15),
      w0, w1, w2, w3);
}

// ---------- kernel 3: pair bias, first streaming pass over pair ----------
__global__ __launch_bounds__(128)
void k_pairbias(const float* __restrict__ pair, const __bf16* __restrict__ wbsw,
                float* __restrict__ pbias) {
  __shared__ __bf16 As[16 * 512];             // 4 chunks x 4 k-tiles, swizzled
  const int tid = threadIdx.x, lane = tid & 31, wv = tid >> 5;
  const size_t base = (size_t)blockIdx.x * 64;  // 64 (n,m) pairs per block
#pragma unroll 4
  for (int it = 0; it < 16; ++it) {           // 2048 float4 loads of pair
    int i = tid + it * 128;
    int r = i >> 5, c4 = (i & 31) * 4;
    float4 s = *(const float4*)(pair + (base + r) * CDIM + c4);
    st4(As + ((r >> 4) * 4 + (c4 >> 5)) * 512 + swzA(r & 15, c4 & 31),
        s.x, s.y, s.z, s.w);
  }
  __syncthreads();
  v8f acc = {};
#pragma unroll
  for (int kt = 0; kt < 4; ++kt)
    acc = wmma_bf16(frag(As + (wv * 4 + kt) * 512 + lane * 16),
                    frag(wbsw + kt * 512 + lane * 16), acc);
  const int half = lane >> 4, hh = lane & 15;
  if (hh < NH) {
#pragma unroll
    for (int r = 0; r < 8; ++r) {
      size_t flat = base + wv * 16 + r + 8 * half;
      pbias[(size_t)hh * NN + flat] = acc[r];
    }
  }
}

// ---------- kernel 4: WMMA logits + fp32 softmax + colsum ----------
__global__ __launch_bounds__(256)
void k_softmax(const __bf16* __restrict__ qsw, const __bf16* __restrict__ ksw,
               const float* __restrict__ pbias, const float* __restrict__ mterm,
               __bf16* __restrict__ wsw, float* __restrict__ colsum) {
  __shared__ float4 Ls4[16 * 128];            // 16 rows x 512 logits
  float* Ls = (float*)Ls4;
  const int tid = threadIdx.x, lane = tid & 31, wv = tid >> 5;
  const int nt = blockIdx.x, n0 = nt * 16, h = blockIdx.y;
  v16bf a = frag(qsw + (size_t)(h * 32 + nt) * 512 + lane * 16);
  const int half = lane >> 4, cn = lane & 15;
  for (int mt = wv; mt < 32; mt += 8) {
    v16bf b = frag(ksw + (size_t)(h * 32 + mt) * 512 + lane * 16);
    v8f z = {};
    v8f acc = wmma_bf16(a, b, z);
#pragma unroll
    for (int r = 0; r < 8; ++r) Ls[(r + 8 * half) * 512 + mt * 16 + cn] = acc[r];
  }
  __syncthreads();
  const size_t pb_base = (size_t)h * NN + (size_t)n0 * NRES;
#pragma unroll
  for (int it = 0; it < 8; ++it) {            // bias + key-term add, float4-wide
    int i4 = tid + it * 256;
    int row = i4 >> 7, mq = (i4 & 127) * 4;
    float4 l = Ls4[i4];
    float4 pb = *(const float4*)(pbias + pb_base + (size_t)row * NRES + mq);
    float4 mm = *(const float4*)(mterm + h * NRES + mq);
    l.x += pb.x + mm.x; l.y += pb.y + mm.y;
    l.z += pb.z + mm.z; l.w += pb.w + mm.w;
    Ls4[i4] = l;
  }
  __syncthreads();
  __bf16* wt = wsw + (size_t)((h * 32 + nt) * 16) * 512;
#pragma unroll
  for (int rr = 0; rr < 2; ++rr) {            // each wave: 2 rows; lane: 16 contiguous keys
    int row = wv * 2 + rr;
    int m0 = lane * 16;
    float v[16];
#pragma unroll
    for (int j = 0; j < 4; ++j) {
      float4 t4 = Ls4[row * 128 + lane * 4 + j];
      v[j * 4 + 0] = t4.x; v[j * 4 + 1] = t4.y;
      v[j * 4 + 2] = t4.z; v[j * 4 + 3] = t4.w;
    }
    float mx = v[0];
#pragma unroll
    for (int j = 1; j < 16; ++j) mx = fmaxf(mx, v[j]);
#pragma unroll
    for (int o = 16; o > 0; o >>= 1) mx = fmaxf(mx, __shfl_xor(mx, o, 32));
    float s = 0.f;
#pragma unroll
    for (int j = 0; j < 16; ++j) { v[j] = __expf(v[j] - mx); s += v[j]; }
#pragma unroll
    for (int o = 16; o > 0; o >>= 1) s += __shfl_xor(s, o, 32);
    float inv = 1.f / s;
#pragma unroll
    for (int j = 0; j < 16; ++j) v[j] *= inv;
#pragma unroll
    for (int j = 0; j < 4; ++j) {             // write back for column sums
      float4 t4; t4.x = v[j * 4]; t4.y = v[j * 4 + 1];
      t4.z = v[j * 4 + 2]; t4.w = v[j * 4 + 3];
      Ls4[row * 128 + lane * 4 + j] = t4;
    }
    st8(wt + (size_t)(m0 >> 5) * 512 + swzA(row, m0 & 31), v);
    st8(wt + (size_t)(m0 >> 5) * 512 + swzA(row, (m0 + 8) & 31), v + 8);
  }
  __syncthreads();
  for (int m = tid; m < 512; m += 256) {      // colsum: plain LDS reads, 1 atomic/col
    float s = 0.f;
#pragma unroll
    for (int r = 0; r < 16; ++r) s += Ls[r * 512 + m];
    atomicAdd(&colsum[h * NRES + m], s);
  }
}

// ---------- kernel 5: value aggregation + inverse frames (no LDS matmul) ----------
__global__ __launch_bounds__(96)
void k_aggregate(const __bf16* __restrict__ wsw, const __bf16* __restrict__ vsw,
                 const float* __restrict__ rot, const float* __restrict__ trans,
                 __bf16* __restrict__ att) {
  __shared__ float Lout[16][24];
  const int tid = threadIdx.x, lane = tid & 31, wv = tid >> 5;
  const int nt = blockIdx.x, n0 = nt * 16, h = blockIdx.y;
  const __bf16* abase = wsw + (size_t)((h * 32 + nt) * 16) * 512 + lane * 16;
  const __bf16* bbase = vsw + ((size_t)(h * 16) * 3 + wv) * 512 + lane * 16;
  v8f acc = {};
#pragma unroll
  for (int kt = 0; kt < 16; ++kt)
    acc = wmma_bf16(frag(abase + (size_t)kt * 512),
                    frag(bbase + (size_t)kt * 3 * 512), acc);
  const int half = lane >> 4, cn = lane & 15;
  const int col = wv * 16 + cn;
#pragma unroll
  for (int r = 0; r < 8; ++r) {
    int row = r + 8 * half;
    float val = acc[r];
    if (col < 16) att[attsw(n0 + row, h * EDIM + col)] = bf(val);
    else if (col < 40) Lout[row][col - 16] = val;
  }
  __syncthreads();
  for (int idx = tid; idx < 128; idx += 96) {     // 16 rows x 8 value points
    int row = idx >> 3, p = idx & 7, n = n0 + row;
    float l0 = Lout[row][p * 3 + 0] - trans[n * 3 + 0];
    float l1 = Lout[row][p * 3 + 1] - trans[n * 3 + 1];
    float l2 = Lout[row][p * 3 + 2] - trans[n * 3 + 2];
    float g[3];
#pragma unroll
    for (int i = 0; i < 3; ++i)                   // g = R^T (l - t)
      g[i] = rot[n * 9 + 0 * 3 + i] * l0 + rot[n * 9 + 1 * 3 + i] * l1 +
             rot[n * 9 + 2 * 3 + i] * l2;
#pragma unroll
    for (int i = 0; i < 3; ++i)
      att[attsw(n, 1728 + p * (NH * 3) + h * 3 + i)] = bf(g[i]);
    float nm = sqrtf(g[0] * g[0] + g[1] * g[1] + g[2] * g[2]);
    att[attsw(n, 2016 + p * NH + h)] = bf(nm);
  }
}

// ---------- kernel 6: pair-value aggregation, second pass over pair ----------
__global__ __launch_bounds__(256)
void k_attpair(const float* __restrict__ pair, const __bf16* __restrict__ colsw,
               __bf16* __restrict__ att) {
  __shared__ __bf16 Bs[8 * 512];                // 8 swizzled column tiles
  const int tid = threadIdx.x, lane = tid & 31, wv = tid >> 5;
  const int i0 = blockIdx.x;
  v8f acc = {};
  for (int kt = 0; kt < 16; ++kt) {
#pragma unroll
    for (int it = 0; it < 4; ++it) {          // 1024 k-quads; c fastest (coalesced)
      int idx = tid + it * 256;
      int cc = idx & 127, b = (idx >> 7) * 4;
      const float* src = pair + ((size_t)i0 * NRES + kt * 32 + b) * CDIM + cc;
      st4(Bs + (cc >> 4) * 512 + swzB(b, cc & 15),
          src[0], src[CDIM], src[2 * CDIM], src[3 * CDIM]);
    }
    __syncthreads();
    acc = wmma_bf16(frag(colsw + kt * 512 + lane * 16),
                    frag(Bs + wv * 512 + lane * 16), acc);
    __syncthreads();
  }
  const int half = lane >> 4, cn = lane & 15;
  const int cc = wv * 16 + cn;
#pragma unroll
  for (int r = 0; r < 8; ++r) {
    int hh = r + 8 * half;
    if (hh < NH) att[attsw(i0, 192 + hh * CDIM + cc)] = bf(acc[r]);
  }
}

// ---------- kernel 7: final projection [512,2112] x [2112,384] (no LDS) ----------
__global__ __launch_bounds__(256)
void k_outproj(const __bf16* __restrict__ att, const __bf16* __restrict__ woutsw,
               const float* __restrict__ bout, float* __restrict__ out) {
  const int tid = threadIdx.x, lane = tid & 31, wv = tid >> 5;
  const int n0 = blockIdx.x * 16;
  const int st = blockIdx.y * 8 + wv;           // 24 s-tiles of 16
  const __bf16* abase = att + (size_t)(blockIdx.x * KT_F) * 512 + lane * 16;
  const __bf16* bbase = woutsw + (size_t)st * 512 + lane * 16;
  v8f acc = {};
#pragma unroll 2
  for (int kt = 0; kt < KT_F; ++kt)
    acc = wmma_bf16(frag(abase + (size_t)kt * 512),
                    frag(bbase + (size_t)kt * 24 * 512), acc);
  const int half = lane >> 4, cn = lane & 15;
  const int s = st * 16 + cn;
#pragma unroll
  for (int r = 0; r < 8; ++r) {
    int row = n0 + r + 8 * half;
    out[(size_t)row * SDIM + s] = acc[r] + bout[s];
  }
}

// ---------- launch ----------
extern "C" void kernel_launch(void* const* d_in, const int* in_sizes, int n_in,
                              void* d_out, int out_size, void* d_ws, size_t ws_size,
                              hipStream_t stream) {
  const float* single = (const float*)d_in[0];
  const float* pair   = (const float*)d_in[1];
  const float* rot    = (const float*)d_in[2];
  const float* trans  = (const float*)d_in[3];
  const float* Wq  = (const float*)d_in[4];
  const float* Wk  = (const float*)d_in[5];
  const float* Wv  = (const float*)d_in[6];
  const float* Wqp = (const float*)d_in[7];
  const float* Wkp = (const float*)d_in[8];
  const float* Wvp = (const float*)d_in[9];
  const float* Wb  = (const float*)d_in[10];
  const float* Wout= (const float*)d_in[11];
  const float* bout= (const float*)d_in[12];
  const float* sc  = (const float*)d_in[13];

  char* ws = (char*)d_ws;
  size_t off = 0;
  auto alloc = [&](size_t bytes) -> void* {
    void* p = ws + off;
    off = (off + bytes + 255) & ~(size_t)255;
    return p;
  };
  float*  qb    = (float*)alloc(sizeof(float) * NH * NRES * EDIM);
  float*  kb    = (float*)alloc(sizeof(float) * NH * NRES * EDIM);
  float*  vb    = (float*)alloc(sizeof(float) * NH * NRES * EDIM);
  float*  qpb   = (float*)alloc(sizeof(float) * NH * NPQ * NRES * 3);
  float*  kpb   = (float*)alloc(sizeof(float) * NH * NPQ * NRES * 3);
  float*  vpb   = (float*)alloc(sizeof(float) * NH * NPV * NRES * 3);
  __bf16* qsw   = (__bf16*)alloc(sizeof(__bf16) * NH * 32 * 512);
  __bf16* ksw   = (__bf16*)alloc(sizeof(__bf16) * NH * 32 * 512);
  float*  mterm = (float*)alloc(sizeof(float) * NH * NRES);
  __bf16* vsw   = (__bf16*)alloc(sizeof(__bf16) * NH * 16 * 3 * 512);
  float*  colsum= (float*)alloc(sizeof(float) * NH * NRES);
  __bf16* colsw = (__bf16*)alloc(sizeof(__bf16) * 16 * 512);
  __bf16* wbsw  = (__bf16*)alloc(sizeof(__bf16) * 4 * 512);
  __bf16* wsw   = (__bf16*)alloc(sizeof(__bf16) * (size_t)NH * 32 * 16 * 512);
  __bf16* attb  = (__bf16*)alloc(sizeof(__bf16) * (size_t)32 * KT_F * 512);
  __bf16* woutsw= (__bf16*)alloc(sizeof(__bf16) * (size_t)KT_F * 24 * 512);
  float*  pbias = (float*)alloc(sizeof(float) * (size_t)NH * NN);

  k_proj<<<dim3(NRES / 16, NH), 192, 0, stream>>>(single, Wq, Wk, Wv, Wqp, Wkp, Wvp,
                                                  qb, kb, vb, qpb, kpb, vpb);
  k_frames<<<(NH * NRES) / 256, 256, 0, stream>>>(rot, trans, sc, qb, kb, vb,
                                                  qpb, kpb, vpb, qsw, ksw,
                                                  mterm, vsw, colsum);
  k_cvt_wb<<<(128 * 16 + 255) / 256, 256, 0, stream>>>(Wb, wbsw);
  k_pairbias<<<NN / 64, 128, 0, stream>>>(pair, wbsw, pbias);
  k_softmax<<<dim3(NRES / 16, NH), 256, 0, stream>>>(qsw, ksw, pbias, mterm,
                                                     wsw, colsum);
  k_cvt_col<<<(16 * 512 + 255) / 256, 256, 0, stream>>>(colsum, colsw);
  k_aggregate<<<dim3(NRES / 16, NH), 96, 0, stream>>>(wsw, vsw, rot, trans, attb);
  k_attpair<<<NRES, 256, 0, stream>>>(pair, colsw, attb);
  k_cvt_wout<<<((FDIM / 4) * SDIM + 255) / 256, 256, 0, stream>>>(Wout, woutsw);
  k_outproj<<<dim3(NRES / 16, SDIM / 128), 256, 0, stream>>>(attb, woutsw, bout,
                                                             (float*)d_out);
}
